// SimpleSSM_29772713295875
// MI455X (gfx1250) — matrix operand order repaired and checked by reading, not compiled, not run on hardware
//
#include <hip/hip_runtime.h>
#include <math.h>

// ---------------------------------------------------------------------------
// SimpleSSM (Mamba-style) for MI455X / gfx1250, wave32, WMMA f16->f32 path.
// v3: multi-buffered B fragments, WGP-scope prefetch, shortened scan chain.
// ---------------------------------------------------------------------------

typedef __attribute__((ext_vector_type(16))) _Float16 v16h;   // 32 bytes
typedef __attribute__((ext_vector_type(8)))  float    v8f;

#define D_MODEL  1024
#define D_STATE  16
#define D_INNER  2048
#define BATCH    4
#define SEQ      2048
#define M_TOTAL  (BATCH * SEQ)   // 8192 rows in all GEMMs
#define DT_CONST 0.1f

__device__ __forceinline__ int lane_id() { return (int)(threadIdx.x & 31u); }

__device__ __forceinline__ float silu_f(float v) {
  return v / (1.0f + __expf(-v));
}

__device__ __forceinline__ v8f wmma_f16(v16h a, v16h b, v8f c) {
  return __builtin_amdgcn_wmma_f32_16x16x32_f16(false, a, false, b,
                                                (short)0, c, false, false);
}

// On-the-fly A fragment (f32 source) -- used only in the skinny B/C GEMM.
__device__ __forceinline__ v16h load_a_f16(const float* __restrict__ A, int lda,
                                           int m0, int k0) {
  const int lane = lane_id();
  const int row  = m0 + (lane & 15);
  const int koff = k0 + ((lane & 16) ? 8 : 0);
  const float* p = A + (size_t)row * (size_t)lda + koff;
  v16h a;
#pragma unroll
  for (int i = 0; i < 8; ++i) {
    a[i]     = (_Float16)p[i];
    a[i + 8] = (_Float16)p[i + 16];
  }
  return a;
}

// ---------------------------------------------------------------------------
// Pack B (K x N f32, row-major) into WMMA B-fragment f16 tiles.
// tile = tn*Ktiles + tk ; lane l: col = tn*16+(l&15), K = tk*32+((l&16)?8:0)+(e<8?e:e+8)
// ---------------------------------------------------------------------------
__global__ void __launch_bounds__(256)
k_pack_b(const float* __restrict__ src, v16h* __restrict__ dst,
         int Ktiles, int N) {
  const int wave = (int)(threadIdx.x >> 5);
  const int lane = lane_id();
  const int tile = (int)blockIdx.x * 8 + wave;
  const int tn = tile / Ktiles;
  const int tk = tile % Ktiles;
  const int col  = tn * 16 + (lane & 15);
  const int koff = tk * 32 + ((lane & 16) ? 8 : 0);
  const float* p = src + (size_t)koff * (size_t)N + col;
  v16h b;
#pragma unroll
  for (int e = 0; e < 16; ++e) b[e] = (_Float16)p[(size_t)(e < 8 ? e : e + 8) * (size_t)N];
  dst[(size_t)tile * 32 + lane] = b;
}

// ---------------------------------------------------------------------------
// Pack A (M x K f32, row-major) into WMMA A-fragment f16 tiles.
// tile = tm*Ktiles + tk ; lane l: row = tm*16+(l&15); K runs {off..off+7},
// {off+16..off+23} with off = tk*32 + ((l&16)?8:0).
// ---------------------------------------------------------------------------
__global__ void __launch_bounds__(256)
k_pack_a(const float* __restrict__ src, v16h* __restrict__ dst, int Ktiles) {
  const int wave = (int)(threadIdx.x >> 5);
  const int lane = lane_id();
  const int tile = (int)blockIdx.x * 8 + wave;
  const int tm = tile / Ktiles;
  const int tk = tile % Ktiles;
  const int row  = tm * 16 + (lane & 15);
  const int koff = tk * 32 + ((lane & 16) ? 8 : 0);
  const int K = Ktiles * 32;
  const float* p = src + (size_t)row * (size_t)K + koff;
  v16h a;
#pragma unroll
  for (int i = 0; i < 8; ++i) {
    a[i]     = (_Float16)p[i];
    a[i + 8] = (_Float16)p[i + 16];
  }
  dst[(size_t)tile * 32 + lane] = a;
}

// ---------------------------------------------------------------------------
// xr = x @ W_in + b_in ; split; SiLU both halves.  Packed operands.
// Block: 8 waves as 2(M) x 4(N); wave tile 32x64; block tile 64x256.
// Grid: (4096/256, 8192/64) = (16, 128).  Ktiles = 32.
// ---------------------------------------------------------------------------
__global__ void __launch_bounds__(256)
k_in_proj(const v16h* __restrict__ Ap, const v16h* __restrict__ Bp,
          const float* __restrict__ b_in, float* __restrict__ xssm,
          float* __restrict__ gate) {
  const int KT = D_MODEL / 32;
  const int wave = (int)(threadIdx.x >> 5);
  const int lane = lane_id();
  const int m0 = (int)blockIdx.y * 64 + (wave >> 2) * 32;
  const int n0 = (int)blockIdx.x * 256 + (wave & 3) * 64;
  const int tm0 = m0 >> 4, tn0 = n0 >> 4;

  v8f acc[2][4] = {};
  for (int tk = 0; tk < KT; ++tk) {
    const v16h a0 = Ap[((size_t)tm0 * KT + tk) * 32 + lane];
    const v16h a1 = Ap[((size_t)(tm0 + 1) * KT + tk) * 32 + lane];
    v16h bfr[4];
#pragma unroll
    for (int j = 0; j < 4; ++j)
      bfr[j] = Bp[((size_t)(tn0 + j) * KT + tk) * 32 + lane];
    if (tk + 1 < KT) {
      __builtin_prefetch((const void*)&Ap[((size_t)tm0 * KT + tk + 1) * 32 + lane], 0, 3);
      __builtin_prefetch((const void*)&Bp[((size_t)tn0 * KT + tk + 1) * 32 + lane], 0, 3);
    }
#pragma unroll
    for (int j = 0; j < 4; ++j) {
      acc[0][j] = wmma_f16(a0, bfr[j], acc[0][j]);
      acc[1][j] = wmma_f16(a1, bfr[j], acc[1][j]);
    }
  }

  const int col = lane & 15;
#pragma unroll
  for (int mi = 0; mi < 2; ++mi) {
    const int rbase = m0 + mi * 16 + ((lane & 16) ? 8 : 0);
#pragma unroll
    for (int j = 0; j < 4; ++j) {
      const int n = n0 + 16 * j + col;
      const float bias = b_in[n];
#pragma unroll
      for (int v = 0; v < 8; ++v) {
        const int m = rbase + v;
        const float val = silu_f(acc[mi][j][v] + bias);
        if (n < D_INNER) xssm[(size_t)m * D_INNER + n] = val;
        else             gate[(size_t)m * D_INNER + (n - D_INNER)] = val;
      }
    }
  }
}

// ---------------------------------------------------------------------------
// Depthwise conv1d, kernel 3, pad 1, along L. One thread per element.
// ---------------------------------------------------------------------------
__global__ void __launch_bounds__(256)
k_dwconv(const float* __restrict__ xssm, const float* __restrict__ conv_w,
         const float* __restrict__ conv_b, float* __restrict__ xconv) {
  const size_t idx = (size_t)blockIdx.x * 256 + threadIdx.x;
  const int d = (int)(idx % D_INNER);
  const size_t ml = idx / D_INNER;         // b*SEQ + l
  const int l = (int)(ml % SEQ);
  const float w0 = conv_w[d * 3 + 0];
  const float w1 = conv_w[d * 3 + 1];
  const float w2 = conv_w[d * 3 + 2];
  const float* base = xssm + ml * (size_t)D_INNER + d;
  float acc = conv_b[d] + w1 * base[0];
  if (l > 0)       acc += w0 * base[-D_INNER];
  if (l < SEQ - 1) acc += w2 * base[D_INNER];
  xconv[ml * (size_t)D_INNER + d] = acc;
}

// ---------------------------------------------------------------------------
// B = xconv @ W_B + b_B + B_mod[batch] ; C likewise.  N = 16 = one tile.
// A converted on the fly (xconv stays f32 for the scan); W_B/W_C packed.
// Grid: 8192/16/8 = 64 blocks.   Ktiles = 64.
// ---------------------------------------------------------------------------
__global__ void __launch_bounds__(256)
k_bc_proj(const float* __restrict__ xconv,
          const v16h* __restrict__ WBp, const v16h* __restrict__ WCp,
          const float* __restrict__ b_B, const float* __restrict__ b_C,
          const float* __restrict__ B_mod, const float* __restrict__ C_mod,
          float* __restrict__ Bout, float* __restrict__ Cout) {
  const int KT = D_INNER / 32;
  const int wave = (int)(threadIdx.x >> 5);
  const int lane = lane_id();
  const int m0 = ((int)blockIdx.x * 8 + wave) * 16;

  v8f accB = {};
  v8f accC = {};
  for (int tk = 0; tk < KT; ++tk) {
    const v16h a  = load_a_f16(xconv, D_INNER, m0, tk * 32);
    const v16h bB = WBp[(size_t)tk * 32 + lane];
    const v16h bC = WCp[(size_t)tk * 32 + lane];
    accB = wmma_f16(a, bB, accB);
    accC = wmma_f16(a, bC, accC);
  }

  const int s     = lane & 15;
  const int rbase = m0 + ((lane & 16) ? 8 : 0);
  const int batch = m0 / SEQ;   // 16-row tile never crosses a batch boundary
  const float addB = b_B[s] + B_mod[batch * D_STATE + s];
  const float addC = b_C[s] + C_mod[batch * D_STATE + s];
#pragma unroll
  for (int v = 0; v < 8; ++v) {
    Bout[(size_t)(rbase + v) * D_STATE + s] = accB[v] + addB;
    Cout[(size_t)(rbase + v) * D_STATE + s] = accC[v] + addC;
  }
}

// ---------------------------------------------------------------------------
// Sequential scan. One thread owns (batch, channel): 16 states in registers.
// B/C (shared by all channels) staged in LDS in chunks of 64 steps.
// y accumulated in 4 partials to shorten the per-step dependency chain.
// Fuses  y = (scan + D*x) * gate.  Grid: BATCH * D_INNER/256 = 32 blocks.
// ---------------------------------------------------------------------------
#define T_CHUNK 64
__global__ void __launch_bounds__(256)
k_scan(const float* __restrict__ xconv, const float* __restrict__ gate,
       const float* __restrict__ Bmat, const float* __restrict__ Cmat,
       const float* __restrict__ A_log, const float* __restrict__ Dvec,
       float* __restrict__ ypre) {
  __shared__ float Bs[T_CHUNK][D_STATE];
  __shared__ float Cs[T_CHUNK][D_STATE];

  const int b = (int)blockIdx.x / (D_INNER / 256);
  const int d = ((int)blockIdx.x % (D_INNER / 256)) * 256 + (int)threadIdx.x;

  float Abar[D_STATE], h[D_STATE];
#pragma unroll
  for (int s = 0; s < D_STATE; ++s) {
    Abar[s] = __expf(-__expf(A_log[d * D_STATE + s]) * DT_CONST);
    h[s] = 0.0f;
  }
  const float Dd = Dvec[d];
  const size_t rowbase = (size_t)b * SEQ;

  for (int t0 = 0; t0 < SEQ; t0 += T_CHUNK) {
    __syncthreads();
    for (int i = (int)threadIdx.x; i < T_CHUNK * D_STATE; i += 256) {
      const int tt = i / D_STATE, ss = i % D_STATE;
      Bs[tt][ss] = Bmat[(rowbase + t0 + tt) * D_STATE + ss];
      Cs[tt][ss] = Cmat[(rowbase + t0 + tt) * D_STATE + ss];
    }
    __syncthreads();
    for (int tt = 0; tt < T_CHUNK; ++tt) {
      const size_t mi = (rowbase + t0 + tt) * (size_t)D_INNER + d;
      const float xv = xconv[mi];
      float y0 = 0.0f, y1 = 0.0f, y2 = 0.0f, y3 = 0.0f;
#pragma unroll
      for (int s = 0; s < D_STATE; s += 4) {
        h[s]     = fmaf(Abar[s],     h[s],     xv * Bs[tt][s]);
        h[s + 1] = fmaf(Abar[s + 1], h[s + 1], xv * Bs[tt][s + 1]);
        h[s + 2] = fmaf(Abar[s + 2], h[s + 2], xv * Bs[tt][s + 2]);
        h[s + 3] = fmaf(Abar[s + 3], h[s + 3], xv * Bs[tt][s + 3]);
        y0 = fmaf(h[s],     Cs[tt][s],     y0);
        y1 = fmaf(h[s + 1], Cs[tt][s + 1], y1);
        y2 = fmaf(h[s + 2], Cs[tt][s + 2], y2);
        y3 = fmaf(h[s + 3], Cs[tt][s + 3], y3);
      }
      const float y = (y0 + y1) + (y2 + y3);
      ypre[mi] = fmaf(Dd, xv, y) * gate[mi];
    }
  }
}

// ---------------------------------------------------------------------------
// out = ypre @ W_out + b_out.  Packed operands, wave tile 32x64.
// Grid: (1024/256, 8192/64) = (4, 128).  Ktiles = 64.
// ---------------------------------------------------------------------------
__global__ void __launch_bounds__(256)
k_out_proj(const v16h* __restrict__ Ap, const v16h* __restrict__ Bp,
           const float* __restrict__ b_out, float* __restrict__ out) {
  const int KT = D_INNER / 32;
  const int wave = (int)(threadIdx.x >> 5);
  const int lane = lane_id();
  const int m0 = (int)blockIdx.y * 64 + (wave >> 2) * 32;
  const int n0 = (int)blockIdx.x * 256 + (wave & 3) * 64;
  const int tm0 = m0 >> 4, tn0 = n0 >> 4;

  v8f acc[2][4] = {};
  for (int tk = 0; tk < KT; ++tk) {
    const v16h a0 = Ap[((size_t)tm0 * KT + tk) * 32 + lane];
    const v16h a1 = Ap[((size_t)(tm0 + 1) * KT + tk) * 32 + lane];
    v16h bfr[4];
#pragma unroll
    for (int j = 0; j < 4; ++j)
      bfr[j] = Bp[((size_t)(tn0 + j) * KT + tk) * 32 + lane];
    if (tk + 1 < KT) {
      __builtin_prefetch((const void*)&Ap[((size_t)tm0 * KT + tk + 1) * 32 + lane], 0, 3);
      __builtin_prefetch((const void*)&Bp[((size_t)tn0 * KT + tk + 1) * 32 + lane], 0, 3);
    }
#pragma unroll
    for (int j = 0; j < 4; ++j) {
      acc[0][j] = wmma_f16(a0, bfr[j], acc[0][j]);
      acc[1][j] = wmma_f16(a1, bfr[j], acc[1][j]);
    }
  }

  const int col = lane & 15;
#pragma unroll
  for (int mi = 0; mi < 2; ++mi) {
    const int rbase = m0 + mi * 16 + ((lane & 16) ? 8 : 0);
#pragma unroll
    for (int j = 0; j < 4; ++j) {
      const int n = n0 + 16 * j + col;
      const float bias = b_out[n];
#pragma unroll
      for (int v = 0; v < 8; ++v)
        out[(size_t)(rbase + v) * D_MODEL + n] = acc[mi][j][v] + bias;
    }
  }
}

// ---------------------------------------------------------------------------
extern "C" void kernel_launch(void* const* d_in, const int* in_sizes, int n_in,
                              void* d_out, int out_size, void* d_ws, size_t ws_size,
                              hipStream_t stream) {
  (void)in_sizes; (void)n_in; (void)out_size; (void)ws_size;

  const float* x      = (const float*)d_in[0];
  const float* B_mod  = (const float*)d_in[1];
  const float* C_mod  = (const float*)d_in[2];
  const float* W_in   = (const float*)d_in[3];
  const float* b_in   = (const float*)d_in[4];
  const float* conv_w = (const float*)d_in[5];
  const float* conv_b = (const float*)d_in[6];
  const float* A_log  = (const float*)d_in[7];
  const float* Dvec   = (const float*)d_in[8];
  const float* W_B    = (const float*)d_in[9];
  const float* b_B    = (const float*)d_in[10];
  const float* W_C    = (const float*)d_in[11];
  const float* b_C    = (const float*)d_in[12];
  const float* W_out  = (const float*)d_in[13];
  const float* b_out  = (const float*)d_in[14];
  float* out = (float*)d_out;

  float* ws = (float*)d_ws;
  const size_t n_md = (size_t)M_TOTAL * D_INNER;   // 16.7M floats

  float* xssm  = ws;               // silu(in_proj half 1); later reused as ypre
  float* gate  = ws + n_md;        // silu(in_proj half 2)
  float* xconv = ws + 2 * n_md;    // depthwise-conv output (f32, scan input)
  float* Bmat  = ws + 3 * n_md;                      // (M_TOTAL, 16)
  float* Cmat  = Bmat + (size_t)M_TOTAL * D_STATE;   // (M_TOTAL, 16)
  float* wpf   = Cmat + (size_t)M_TOTAL * D_STATE;
  v16h* Wp_in  = (v16h*)wpf;                                   // 8 MB
  v16h* Wp_out = (v16h*)(wpf + (size_t)D_MODEL * 2 * D_INNER / 2);
  v16h* WBp    = (v16h*)(wpf + (size_t)D_MODEL * 2 * D_INNER / 2
                             + (size_t)D_INNER * D_MODEL / 2);
  v16h* WCp    = WBp + (size_t)(D_INNER / 32) * 32;
  float* ypre  = xssm;             // xssm dead after conv
  // Packed activations live in the xconv region while xconv is dead:
  //   Xp (16 MB) before xconv is written; Yp (32 MB) after the scan.
  v16h* Xp = (v16h*)xconv;
  v16h* Yp = (v16h*)xconv;

  // ---- pack weights (one-time per launch) ----
  k_pack_b<<<dim3(1024), 256, 0, stream>>>(W_in,  Wp_in,  D_MODEL / 32, 2 * D_INNER);
  k_pack_b<<<dim3(512),  256, 0, stream>>>(W_out, Wp_out, D_INNER / 32, D_MODEL);
  k_pack_b<<<dim3(8),    256, 0, stream>>>(W_B,   WBp,    D_INNER / 32, D_STATE);
  k_pack_b<<<dim3(8),    256, 0, stream>>>(W_C,   WCp,    D_INNER / 32, D_STATE);

  // ---- pipeline ----
  k_pack_a <<<dim3(2048), 256, 0, stream>>>(x, Xp, D_MODEL / 32);
  k_in_proj<<<dim3(16, 128), 256, 0, stream>>>(Xp, Wp_in, b_in, xssm, gate);
  k_dwconv <<<dim3((unsigned)(n_md / 256)), 256, 0, stream>>>(xssm, conv_w, conv_b, xconv);
  k_bc_proj<<<dim3(64), 256, 0, stream>>>(xconv, WBp, WCp, b_B, b_C,
                                          B_mod, C_mod, Bmat, Cmat);
  k_scan   <<<dim3(32), 256, 0, stream>>>(xconv, gate, Bmat, Cmat, A_log, Dvec, ypre);
  k_pack_a <<<dim3(4096), 256, 0, stream>>>(ypre, Yp, D_INNER / 32);
  k_out_proj<<<dim3(4, 128), 256, 0, stream>>>(Yp, Wp_out, b_out, out);
}
